// RetrospectiveTextDecoder_65180423684237
// MI455X (gfx1250) — matrix operand order repaired and checked by reading, not compiled, not run on hardware
//
#include <hip/hip_runtime.h>
#include <hip/hip_bf16.h>
#include <math.h>

// ---- problem constants (match reference) ----
#define B_   8
#define S_   512
#define D_   1024
#define L_   6
#define H_   8
#define HD_  128
#define FF_  4096
#define M_   (B_*S_)          // 4096 token rows
#define EPS_ 1e-5f
#define SCALE_ 0.08838834764831845f   // 1/sqrt(128)

typedef __bf16 bf16;
typedef __attribute__((ext_vector_type(16))) bf16  v16bf;
typedef __attribute__((ext_vector_type(8)))  bf16  v8bf;
typedef __attribute__((ext_vector_type(8)))  float v8f;

#ifndef __has_builtin
#define __has_builtin(x) 0
#endif
#if __has_builtin(__builtin_amdgcn_global_load_async_to_lds_b128) && \
    __has_builtin(__builtin_amdgcn_s_wait_asynccnt)
#define USE_ASYNC_LDS 1
// b128 async copy payload type: int32x4, with explicit address spaces
typedef int i32x4 __attribute__((vector_size(16)));
typedef __attribute__((address_space(1))) i32x4 gbl_i32x4;
typedef __attribute__((address_space(3))) i32x4 lds_i32x4;
#else
#define USE_ASYNC_LDS 0
#endif

// ---------------------------------------------------------------------------
// Weight transpose + fp32->bf16 convert:  W[K,N] (row-major) -> Wt[N,K]
// ---------------------------------------------------------------------------
__global__ __launch_bounds__(256) void convT_kernel(const float* __restrict__ W,
                                                    bf16* __restrict__ Wt,
                                                    int K, int N)
{
    size_t i = (size_t)blockIdx.x * 256 + threadIdx.x;
    size_t total = (size_t)K * N;
    if (i >= total) return;
    int n = (int)(i / K);
    int k = (int)(i % K);
    Wt[i] = (bf16)W[(size_t)k * N + n];
}

// ---------------------------------------------------------------------------
// Embedding: h[b,s,:] = tok_emb[ids[b,s],:] + pos_emb[pos0+s,:]
// ---------------------------------------------------------------------------
__global__ __launch_bounds__(256) void embed_kernel(const int* __restrict__ ids,
                                                    const int* __restrict__ curpos,
                                                    const float* __restrict__ tok,
                                                    const float* __restrict__ pos,
                                                    float* __restrict__ h)
{
    size_t i = (size_t)blockIdx.x * 256 + threadIdx.x;   // over M_*D_
    int d = (int)(i & (D_ - 1));
    int t = (int)(i >> 10);                 // token index, D_=1024
    int s = t & (S_ - 1);
    int id = ids[t];
    h[i] = tok[(size_t)id * D_ + d] + pos[(size_t)(curpos[0] + s) * D_ + d];
}

// ---------------------------------------------------------------------------
// LayerNorm (per token row) -> bf16 output
// ---------------------------------------------------------------------------
__global__ __launch_bounds__(256) void ln_bf16_kernel(const float* __restrict__ x,
                                                      const float* __restrict__ g,
                                                      const float* __restrict__ bta,
                                                      bf16* __restrict__ out)
{
    const int t = blockIdx.x;
    const float* xr = x + (size_t)t * D_;
    float s = 0.f, s2 = 0.f;
    for (int d = threadIdx.x; d < D_; d += 256) {
        float v = xr[d];
        s += v; s2 += v * v;
    }
#pragma unroll
    for (int o = 16; o > 0; o >>= 1) {
        s  += __shfl_xor(s,  o, 32);
        s2 += __shfl_xor(s2, o, 32);
    }
    __shared__ float shs[8], shs2[8];
    int w = threadIdx.x >> 5, lane = threadIdx.x & 31;
    if (lane == 0) { shs[w] = s; shs2[w] = s2; }
    __syncthreads();
    if (threadIdx.x == 0) {
        float a = 0.f, c = 0.f;
#pragma unroll
        for (int i = 0; i < 8; ++i) { a += shs[i]; c += shs2[i]; }
        shs[0] = a; shs2[0] = c;
    }
    __syncthreads();
    float mean = shs[0] * (1.0f / D_);
    float var  = shs2[0] * (1.0f / D_) - mean * mean;
    float rstd = rsqrtf(var + EPS_);
    for (int d = threadIdx.x; d < D_; d += 256) {
        float v = (xr[d] - mean) * rstd * g[d] + bta[d];
        out[(size_t)t * D_ + d] = (bf16)v;
    }
}

// ---------------------------------------------------------------------------
// WMMA GEMM:  C[M,N] = A[M,K](bf16) @ Wt[N,K](bf16)^T + bias
//   Block (4 waves) computes 64M x 128N. Wave tile: 32M x 64N
//   (2 A-frags x 4 B-frags = 8 wmma per K-step of 32).
//   A tile (64 x 64 bf16) double-buffer staged in LDS via
//   GLOBAL_LOAD_ASYNC_TO_LDS_B128 (ASYNCcnt) when available.
//   mode 0: outF[m*N+n] = val
//   mode 1: QKV scatter -> outF as [B,H,S,HD]
//   mode 2: resid[m*N+n] += val
//   mode 3: outB[m*N+n] = bf16(gelu_exact(val))
// ---------------------------------------------------------------------------
#define KC 64

__device__ __forceinline__ void stage_tile(const bf16* __restrict__ A,
                                           bf16* __restrict__ sbuf,
                                           int tid, int m0, int ks, int K)
{
#pragma unroll
    for (int c = 0; c < 4; ++c) {
        int e   = (c * 128 + tid) * 8;     // element index in 64x64 tile
        int row = e >> 6;                  // / KC
        int col = e & (KC - 1);
        const bf16* gp = A + (size_t)(m0 + row) * K + ks + col;
        bf16* lp = sbuf + e;
#if USE_ASYNC_LDS
        __builtin_amdgcn_global_load_async_to_lds_b128(
            (gbl_i32x4*)gp, (lds_i32x4*)lp, 0, 0);
#else
        *(v8bf*)lp = *(const v8bf*)gp;
#endif
    }
}

__global__ __launch_bounds__(128) void gemm_wmma_kernel(const bf16* __restrict__ A,
                                                        const bf16* __restrict__ Wt,
                                                        const float* __restrict__ bias,
                                                        float* __restrict__ outF,
                                                        bf16* __restrict__ outB,
                                                        float* __restrict__ resid,
                                                        int N, int K, int mode)
{
    __shared__ bf16 sA[2][64 * KC];        // 2 x 8KB double buffer

    const int tid  = threadIdx.x;
    const int lane = tid & 31;
    const int wv   = tid >> 5;
    const int sel  = lane >> 4;            // 0: lanes 0-15, 1: lanes 16-31
    const int l16  = lane & 15;
    const int m0   = blockIdx.x * 64;
    const int n0   = blockIdx.y * 128;
    const int mSub = (wv & 1) * 32;        // wave's M offset inside block tile
    const int nSub = (wv >> 1) * 64;       // wave's N offset inside block tile

    v8f acc[2][4];
#pragma unroll
    for (int mi = 0; mi < 2; ++mi)
#pragma unroll
        for (int t = 0; t < 4; ++t)
            acc[mi][t] = (v8f){0.f, 0.f, 0.f, 0.f, 0.f, 0.f, 0.f, 0.f};

    const bf16* Bbase = Wt + (size_t)(n0 + nSub + l16) * K;

    int cur = 0;
    stage_tile(A, &sA[0][0], tid, m0, 0, K);
#if USE_ASYNC_LDS
    __builtin_amdgcn_s_wait_asynccnt(0);
#endif
    __syncthreads();

    for (int ks = 0; ks < K; ks += KC) {
        if (ks + KC < K)
            stage_tile(A, &sA[cur ^ 1][0], tid, m0, ks + KC, K);

#pragma unroll
        for (int kk = 0; kk < KC; kk += 32) {
            // A fragments from LDS (contiguous k = 16*sel + s, matching B)
            v16bf a[2];
#pragma unroll
            for (int mi = 0; mi < 2; ++mi) {
                const bf16* base = &sA[cur][(mSub + mi * 16 + l16) * KC + kk];
                v8bf lo = *(const v8bf*)(base + sel * 16);
                v8bf hi = *(const v8bf*)(base + sel * 16 + 8);
#pragma unroll
                for (int i = 0; i < 8; ++i) { a[mi][i] = lo[i]; a[mi][i + 8] = hi[i]; }
            }
            // B fragments from global (L2-resident weights)
            const bf16* bb = Bbase + ks + kk + sel * 16;
#pragma unroll
            for (int t = 0; t < 4; ++t) {
                v16bf b = *(const v16bf*)(bb + (size_t)t * 16 * K);
                acc[0][t] = __builtin_amdgcn_wmma_f32_16x16x32_bf16(false, a[0], false, b, (short)0, acc[0][t], false, false);
                acc[1][t] = __builtin_amdgcn_wmma_f32_16x16x32_bf16(false, a[1], false, b, (short)0, acc[1][t], false, false);
            }
        }

#if USE_ASYNC_LDS
        __builtin_amdgcn_s_wait_asynccnt(0);
#endif
        __syncthreads();
        cur ^= 1;
    }

    // Epilogue. C/D layout: row = sel*8 + r, col = lane&15.
#pragma unroll
    for (int mi = 0; mi < 2; ++mi) {
#pragma unroll
        for (int t = 0; t < 4; ++t) {
            int n = n0 + nSub + t * 16 + l16;
            float bvv = bias[n];
#pragma unroll
            for (int r = 0; r < 8; ++r) {
                int m = m0 + mSub + mi * 16 + sel * 8 + r;
                float val = acc[mi][t][r] + bvv;
                if (mode == 0) {
                    outF[(size_t)m * N + n] = val;
                } else if (mode == 1) {
                    int bb = m >> 9;              // m / S_
                    int ss = m & (S_ - 1);
                    int hh = n >> 7;              // n / HD_
                    int hd = n & (HD_ - 1);
                    outF[(((size_t)(bb * H_ + hh)) * S_ + ss) * HD_ + hd] = val;
                } else if (mode == 2) {
                    resid[(size_t)m * N + n] += val;
                } else {
                    float gv = 0.5f * val * (1.f + erff(val * 0.70710678118654752f));
                    outB[(size_t)m * N + n] = (bf16)gv;
                }
            }
        }
    }
}

// ---------------------------------------------------------------------------
// Attention: one wave per (b,h,query). Two-pass softmax with LDS score buffer.
// q,k,v are fp32 [B,H,S,HD]; output written as bf16 [B,S,D] (GEMM-ready).
// ---------------------------------------------------------------------------
__global__ __launch_bounds__(128) void attn_kernel(const float* __restrict__ qg,
                                                   const float* __restrict__ kg,
                                                   const float* __restrict__ vg,
                                                   const int* __restrict__ amask,
                                                   bf16* __restrict__ ob)
{
    __shared__ float sQ[4][HD_];
    __shared__ float sS[4][S_];
    const int lane = threadIdx.x & 31;
    const int w    = threadIdx.x >> 5;
    const int qi   = blockIdx.x * 4 + w;       // 0 .. B*H*S-1
    const int s  = qi & (S_ - 1);
    const int bh = qi >> 9;                    // / S_
    const int hh = bh & (H_ - 1);
    const int b  = bh >> 3;                    // / H_

    const float* qr = qg + ((size_t)(b * H_ + hh) * S_ + s) * HD_;
    for (int d = lane; d < HD_; d += 32) sQ[w][d] = qr[d];
    __syncthreads();

    // pass 1: scores + running max (vectorized b128 K-row loads)
    const float* kb = kg + (size_t)(b * H_ + hh) * S_ * HD_;
    float mx = -3.0e38f;
    for (int j = lane; j < S_; j += 32) {
        const float4* kr4 = (const float4*)(kb + (size_t)j * HD_);
        float dot = 0.f;
#pragma unroll 8
        for (int d4 = 0; d4 < HD_ / 4; ++d4) {
            float4 kv = kr4[d4];
            dot += sQ[w][4 * d4 + 0] * kv.x;
            dot += sQ[w][4 * d4 + 1] * kv.y;
            dot += sQ[w][4 * d4 + 2] * kv.z;
            dot += sQ[w][4 * d4 + 3] * kv.w;
        }
        dot *= SCALE_;
        if (amask[b * S_ + j] == 0) dot = -10000.f;
        sS[w][j] = dot;
        mx = fmaxf(mx, dot);
    }
#pragma unroll
    for (int o = 16; o > 0; o >>= 1) mx = fmaxf(mx, __shfl_xor(mx, o, 32));

    float sum = 0.f;
    for (int j = lane; j < S_; j += 32) {
        float e = __expf(sS[w][j] - mx);
        sS[w][j] = e;
        sum += e;
    }
#pragma unroll
    for (int o = 16; o > 0; o >>= 1) sum += __shfl_xor(sum, o, 32);
    float inv = 1.f / sum;
    __syncthreads();

    // pass 2: o[d] = sum_j p_j * v[j,d]; lane owns 4 contiguous d
    const float* vb = vg + (size_t)(b * H_ + hh) * S_ * HD_;
    const int d0 = lane * 4;
    float a0 = 0.f, a1 = 0.f, a2 = 0.f, a3 = 0.f;
    for (int j = 0; j < S_; ++j) {
        float wt = sS[w][j];
        const float4 vr = *(const float4*)(vb + (size_t)j * HD_ + d0);
        a0 += wt * vr.x;
        a1 += wt * vr.y;
        a2 += wt * vr.z;
        a3 += wt * vr.w;
    }
    bf16* orow = ob + ((size_t)(b * S_ + s)) * D_ + hh * HD_ + d0;
    orow[0] = (bf16)(a0 * inv);
    orow[1] = (bf16)(a1 * inv);
    orow[2] = (bf16)(a2 * inv);
    orow[3] = (bf16)(a3 * inv);
}

// ---------------------------------------------------------------------------
extern "C" void kernel_launch(void* const* d_in, const int* in_sizes, int n_in,
                              void* d_out, int out_size, void* d_ws, size_t ws_size,
                              hipStream_t stream)
{
    (void)in_sizes; (void)n_in; (void)out_size; (void)ws_size;

    const int*   ids   = (const int*)d_in[0];
    const int*   amask = (const int*)d_in[1];
    const int*   curp  = (const int*)d_in[2];
    const float* tok   = (const float*)d_in[3];
    const float* pose  = (const float*)d_in[4];
    const float* ln1s  = (const float*)d_in[5];
    const float* ln1b  = (const float*)d_in[6];
    const float* Wq    = (const float*)d_in[7];
    const float* bq    = (const float*)d_in[8];
    const float* Wk    = (const float*)d_in[9];
    const float* bk    = (const float*)d_in[10];
    const float* Wv    = (const float*)d_in[11];
    const float* bvp   = (const float*)d_in[12];
    const float* Wo    = (const float*)d_in[13];
    const float* bo    = (const float*)d_in[14];
    const float* ln2s  = (const float*)d_in[15];
    const float* ln2b  = (const float*)d_in[16];
    const float* W1    = (const float*)d_in[17];
    const float* b1    = (const float*)d_in[18];
    const float* W2    = (const float*)d_in[19];
    const float* b2    = (const float*)d_in[20];
    const float* lnfs  = (const float*)d_in[21];
    const float* lnfb  = (const float*)d_in[22];
    const float* Wout  = (const float*)d_in[23];
    const float* bout  = (const float*)d_in[24];
    float* out = (float*)d_out;

    // ---- workspace carve ----
    size_t off = 0;
    auto carve = [&](size_t bytes) -> void* {
        void* p = (char*)d_ws + off;
        off = (off + bytes + 255) & ~(size_t)255;
        return p;
    };
    const size_t DD  = (size_t)D_ * D_;
    const size_t DFF = (size_t)D_ * FF_;
    bf16* WqT   = (bf16*)carve((size_t)L_ * DD  * 2);
    bf16* WkT   = (bf16*)carve((size_t)L_ * DD  * 2);
    bf16* WvT   = (bf16*)carve((size_t)L_ * DD  * 2);
    bf16* WoT   = (bf16*)carve((size_t)L_ * DD  * 2);
    bf16* W1T   = (bf16*)carve((size_t)L_ * DFF * 2);
    bf16* W2T   = (bf16*)carve((size_t)L_ * DFF * 2);
    bf16* WoutT = (bf16*)carve(DD * 2);
    float* h    = (float*)carve((size_t)M_ * D_ * 4);
    bf16*  xb   = (bf16*)carve((size_t)M_ * D_ * 2);
    float* qB   = (float*)carve((size_t)M_ * D_ * 4);
    float* kB   = (float*)carve((size_t)M_ * D_ * 4);
    float* vB   = (float*)carve((size_t)M_ * D_ * 4);
    bf16*  ob   = (bf16*)carve((size_t)M_ * D_ * 2);
    bf16*  m1b  = (bf16*)carve((size_t)M_ * FF_ * 2);

    // ---- 1) weight convert/transpose (fp32 -> bf16 [N,K]) ----
    const int gDDc  = (int)((DD  + 255) / 256);
    const int gDFFc = (int)((DFF + 255) / 256);
    for (int l = 0; l < L_; ++l) {
        convT_kernel<<<gDDc,  256, 0, stream>>>(Wq + (size_t)l * DD,  WqT + (size_t)l * DD,  D_,  D_);
        convT_kernel<<<gDDc,  256, 0, stream>>>(Wk + (size_t)l * DD,  WkT + (size_t)l * DD,  D_,  D_);
        convT_kernel<<<gDDc,  256, 0, stream>>>(Wv + (size_t)l * DD,  WvT + (size_t)l * DD,  D_,  D_);
        convT_kernel<<<gDDc,  256, 0, stream>>>(Wo + (size_t)l * DD,  WoT + (size_t)l * DD,  D_,  D_);
        convT_kernel<<<gDFFc, 256, 0, stream>>>(W1 + (size_t)l * DFF, W1T + (size_t)l * DFF, D_,  FF_);
        convT_kernel<<<gDFFc, 256, 0, stream>>>(W2 + (size_t)l * DFF, W2T + (size_t)l * DFF, FF_, D_);
    }
    convT_kernel<<<gDDc, 256, 0, stream>>>(Wout, WoutT, D_, D_);

    // ---- 2) embedding ----
    embed_kernel<<<(M_ * D_) / 256, 256, 0, stream>>>(ids, curp, tok, pose, h);

    // ---- 3) layers ----
    const dim3 gD (M_ / 64, D_  / 128);   // GEMM N=1024: (64, 8)
    const dim3 gFF(M_ / 64, FF_ / 128);   // GEMM N=4096: (64, 32)
    for (int l = 0; l < L_; ++l) {
        ln_bf16_kernel<<<M_, 256, 0, stream>>>(h, ln1s + (size_t)l * D_, ln1b + (size_t)l * D_, xb);

        gemm_wmma_kernel<<<gD, 128, 0, stream>>>(xb, WqT + (size_t)l * DD, bq  + (size_t)l * D_,
                                                 qB, nullptr, nullptr, D_, D_, 1);
        gemm_wmma_kernel<<<gD, 128, 0, stream>>>(xb, WkT + (size_t)l * DD, bk  + (size_t)l * D_,
                                                 kB, nullptr, nullptr, D_, D_, 1);
        gemm_wmma_kernel<<<gD, 128, 0, stream>>>(xb, WvT + (size_t)l * DD, bvp + (size_t)l * D_,
                                                 vB, nullptr, nullptr, D_, D_, 1);

        attn_kernel<<<(B_ * H_ * S_) / 4, 128, 0, stream>>>(qB, kB, vB, amask, ob);

        gemm_wmma_kernel<<<gD, 128, 0, stream>>>(ob, WoT + (size_t)l * DD, bo + (size_t)l * D_,
                                                 nullptr, nullptr, h, D_, D_, 2);

        ln_bf16_kernel<<<M_, 256, 0, stream>>>(h, ln2s + (size_t)l * D_, ln2b + (size_t)l * D_, xb);

        gemm_wmma_kernel<<<gFF, 128, 0, stream>>>(xb, W1T + (size_t)l * DFF, b1 + (size_t)l * FF_,
                                                  nullptr, m1b, nullptr, FF_, D_, 3);
        gemm_wmma_kernel<<<gD, 128, 0, stream>>>(m1b, W2T + (size_t)l * DFF, b2 + (size_t)l * D_,
                                                 nullptr, nullptr, h, D_, FF_, 2);
    }

    // ---- 4) final LN + output projection ----
    ln_bf16_kernel<<<M_, 256, 0, stream>>>(h, lnfs, lnfb, xb);
    gemm_wmma_kernel<<<gD, 128, 0, stream>>>(xb, WoutT, bout, out, nullptr, nullptr, D_, D_, 0);
}
